// Mesh2GridDecoder_11991548690709
// MI455X (gfx1250) — matrix operand order repaired and checked by reading, not compiled, not run on hardware
//
#include <hip/hip_runtime.h>
#include <stdint.h>

#define N_MESH  10242
#define N_GRIDN 65160
#define N_EDGES 195480
#define DD      512
#define NOUT    471
#define A_STRIDE 520   // padded LDS row stride (bf16 elems): 1040B, 16B-aligned, bank-staggered

typedef __attribute__((ext_vector_type(16))) __bf16 v16bf;
typedef __attribute__((ext_vector_type(8)))  float  v8f;

__device__ __forceinline__ unsigned short f2bf(float x) {
  unsigned u = __float_as_uint(x);
  u += 0x7fffu + ((u >> 16) & 1u);        // round-to-nearest-even
  return (unsigned short)(u >> 16);
}

__device__ __forceinline__ float silu_f(float x) {
  return x / (1.0f + __expf(-x));
}

__device__ __forceinline__ v16bf load_a_frag(const unsigned short* p) {
  union { v16bf v; uint4 q[2]; } u;
  u.q[0] = *(const uint4*)(p);        // K = base .. base+7
  u.q[1] = *(const uint4*)(p + 16);   // K = base+16 .. base+23
  return u.v;
}

__device__ __forceinline__ v16bf load_b_frag(const unsigned short* P, long f, int lane) {
  const unsigned short* p = P + ((size_t)f * 32 + (size_t)lane) * 16;
  union { v16bf v; uint4 q[2]; } u;
  u.q[0] = *(const uint4*)(p);
  u.q[1] = *(const uint4*)(p + 8);
  return u.v;
}

__device__ __forceinline__ v8f wmma_bf16(v16bf a, v16bf b, v8f c) {
  return __builtin_amdgcn_wmma_f32_16x16x32_bf16(false, a, false, b, (short)0, c, false, false);
}

__device__ __forceinline__ v8f splat8(float v) {
  v8f r = {v, v, v, v, v, v, v, v};
  return r;
}

// ---------------- f32 -> bf16 cast ----------------
__global__ void k_cast(const float* __restrict__ src, unsigned short* __restrict__ dst, size_t n) {
  size_t i = (size_t)blockIdx.x * blockDim.x + threadIdx.x;
  size_t stride = (size_t)gridDim.x * blockDim.x;
  for (; i < n; i += stride) dst[i] = f2bf(src[i]);
}

// ------- pack fp32 row-major W[K][N] into WMMA B-fragments (bf16) -------
// frag f = nt*Kt + kt ; within frag: lane l holds 16 contiguous bf16:
//   n = nt*16 + (l&15), k = kt*32 + (l>>4)*8 + (j<8 ? j : j+8)   (j = 0..15)
__global__ void k_pack(const float* __restrict__ W, unsigned short* __restrict__ P,
                       int K, int Nsrc, int Nt) {
  int t = blockIdx.x * blockDim.x + threadIdx.x;
  int Kt = K >> 5;
  if (t >= Nt * Kt * 32) return;
  int f = t >> 5, lane = t & 31;
  int nt = f / Kt, kt = f % Kt;
  int n = nt * 16 + (lane & 15);
  int kbase = kt * 32 + (lane >> 4) * 8;
  unsigned short* dst = P + ((size_t)f * 32 + lane) * 16;
#pragma unroll
  for (int j = 0; j < 16; ++j) {
    float v = (n < Nsrc) ? W[(size_t)(kbase + (j < 8 ? j : j + 8)) * Nsrc + n] : 0.0f;
    dst[j] = f2bf(v);
  }
}

// ---------------- fused per-16-edge-tile pipeline ----------------
__global__ __launch_bounds__(256) void k_edge(
    const unsigned short* __restrict__ mesh_bf,
    const unsigned short* __restrict__ grid_bf,
    const float* __restrict__ edge_attrs,
    const int* __restrict__ edge_src,
    const int* __restrict__ edge_dst,
    const float* __restrict__ emb_w0, const float* __restrict__ emb_b0,
    const unsigned short* __restrict__ P_emb_w1, const float* __restrict__ emb_b1,
    const unsigned short* __restrict__ P_edge_w0, const float* __restrict__ edge_b0,
    const unsigned short* __restrict__ P_edge_w1, const float* __restrict__ edge_b1,
    float* __restrict__ agg)
{
  __shared__ __align__(16) unsigned short bufA[16 * A_STRIDE]; // src rows -> hidden act t
  __shared__ __align__(16) unsigned short bufB[16 * A_STRIDE]; // dst rows
  __shared__ __align__(16) unsigned short bufC[16 * A_STRIDE]; // emb hidden h -> e
  __shared__ int s_dst[16];

  const int tid  = threadIdx.x;
  const int w    = tid >> 5;
  const int lane = tid & 31;
  const int nb   = lane & 15;
  const int row0 = blockIdx.x * 16;

  if (tid < 16) {
    int e = row0 + tid; if (e > N_EDGES - 1) e = N_EDGES - 1;
    s_dst[tid] = edge_dst[e];
  }
  // gather src/dst rows (bf16, 1KB per row) cooperatively
  for (int i = tid; i < 16 * 64; i += 256) {
    int r = i >> 6, c = i & 63;
    int e = row0 + r; if (e > N_EDGES - 1) e = N_EDGES - 1;
    int si = edge_src[e];
    int di = edge_dst[e];
    *(uint4*)(&bufA[r * A_STRIDE + c * 8]) = *(const uint4*)(&mesh_bf[(size_t)si * DD + (size_t)c * 8]);
    *(uint4*)(&bufB[r * A_STRIDE + c * 8]) = *(const uint4*)(&grid_bf[(size_t)di * DD + (size_t)c * 8]);
  }
  // edge-embedding layer 0 (K=4, VALU): h = silu(attr @ emb_w0 + b0)
  for (int i = tid; i < 16 * DD; i += 256) {
    int r = i >> 9, c = i & (DD - 1);
    int e = row0 + r; if (e > N_EDGES - 1) e = N_EDGES - 1;
    const float* a4 = edge_attrs + (size_t)e * 4;
    float s = emb_b0[c] + a4[0] * emb_w0[c] + a4[1] * emb_w0[DD + c]
            + a4[2] * emb_w0[2 * DD + c] + a4[3] * emb_w0[3 * DD + c];
    bufC[r * A_STRIDE + c] = f2bf(silu_f(s));
  }
  __syncthreads();

  // ---- emb layer 1: e = h @ emb_w1 + b1 ----
  v8f Ce[4];
#pragma unroll
  for (int j = 0; j < 4; ++j) Ce[j] = splat8(emb_b1[(4 * w + j) * 16 + nb]);
  for (int kt = 0; kt < 16; ++kt) {
    v16bf a = load_a_frag(&bufC[nb * A_STRIDE + kt * 32 + (lane >> 4) * 8]);
#pragma unroll
    for (int j = 0; j < 4; ++j)
      Ce[j] = wmma_bf16(a, load_b_frag(P_emb_w1, (long)(4 * w + j) * 16 + kt, lane), Ce[j]);
  }
  __syncthreads();
  // write e (bf16) over bufC; keep f32 copy in Ce for the residual
#pragma unroll
  for (int j = 0; j < 4; ++j) {
    int n = (4 * w + j) * 16 + nb;
#pragma unroll
    for (int r = 0; r < 8; ++r) {
      int m = r + ((lane >> 4) << 3);
      bufC[m * A_STRIDE + n] = f2bf(Ce[j][r]);
    }
  }
  __syncthreads();

  // ---- edge MLP hidden: t = silu([src|dst|e] @ edge_w0 + b0), 3 accumulating GEMMs ----
  v8f Ch[4];
#pragma unroll
  for (int j = 0; j < 4; ++j) Ch[j] = splat8(edge_b0[(4 * w + j) * 16 + nb]);
  for (int kt = 0; kt < 16; ++kt) {
    const int aoff = nb * A_STRIDE + kt * 32 + (lane >> 4) * 8;
    v16bf as = load_a_frag(&bufA[aoff]);
    v16bf ad = load_a_frag(&bufB[aoff]);
    v16bf ae = load_a_frag(&bufC[aoff]);
#pragma unroll
    for (int j = 0; j < 4; ++j) {
      long fb = (long)(4 * w + j) * 48;   // edge_w0 packed with Kt = 48
      Ch[j] = wmma_bf16(as, load_b_frag(P_edge_w0, fb + kt,      lane), Ch[j]);
      Ch[j] = wmma_bf16(ad, load_b_frag(P_edge_w0, fb + 16 + kt, lane), Ch[j]);
      Ch[j] = wmma_bf16(ae, load_b_frag(P_edge_w0, fb + 32 + kt, lane), Ch[j]);
    }
  }
  __syncthreads();
#pragma unroll
  for (int j = 0; j < 4; ++j) {
    int n = (4 * w + j) * 16 + nb;
#pragma unroll
    for (int r = 0; r < 8; ++r) {
      int m = r + ((lane >> 4) << 3);
      bufA[m * A_STRIDE + n] = f2bf(silu_f(Ch[j][r]));
    }
  }
  __syncthreads();

  // ---- edge MLP out + residual, scatter-add into agg ----
  v8f Co[4];
#pragma unroll
  for (int j = 0; j < 4; ++j) Co[j] = splat8(edge_b1[(4 * w + j) * 16 + nb]);
  for (int kt = 0; kt < 16; ++kt) {
    v16bf a = load_a_frag(&bufA[nb * A_STRIDE + kt * 32 + (lane >> 4) * 8]);
#pragma unroll
    for (int j = 0; j < 4; ++j)
      Co[j] = wmma_bf16(a, load_b_frag(P_edge_w1, (long)(4 * w + j) * 16 + kt, lane), Co[j]);
  }
  int nvalid = N_EDGES - row0; if (nvalid > 16) nvalid = 16;
#pragma unroll
  for (int j = 0; j < 4; ++j) {
    int n = (4 * w + j) * 16 + nb;
#pragma unroll
    for (int r = 0; r < 8; ++r) {
      int m = r + ((lane >> 4) << 3);
      if (m < nvalid)
        atomicAdd(&agg[(size_t)s_dst[m] * DD + n], Ce[j][r] + Co[j][r]);
    }
  }
}

// ------------- fused node MLP + output MLP per 16-grid-row tile -------------
__global__ __launch_bounds__(256) void k_node(
    const unsigned short* __restrict__ grid_bf,
    const float* __restrict__ grid_f32,
    const float* __restrict__ agg,
    const unsigned short* __restrict__ P_node_w0, const float* __restrict__ node_b0,
    const unsigned short* __restrict__ P_node_w1, const float* __restrict__ node_b1,
    const unsigned short* __restrict__ P_out_w0,  const float* __restrict__ out_b0,
    const unsigned short* __restrict__ P_out_w1,  const float* __restrict__ out_b1,
    float* __restrict__ out)
{
  __shared__ __align__(16) unsigned short bufA[16 * A_STRIDE]; // grid rows -> t1 -> t2
  __shared__ __align__(16) unsigned short bufB[16 * A_STRIDE]; // agg rows -> grid_new

  const int tid  = threadIdx.x;
  const int w    = tid >> 5;
  const int lane = tid & 31;
  const int nb   = lane & 15;
  const int row0 = blockIdx.x * 16;

  for (int i = tid; i < 16 * 64; i += 256) {
    int r = i >> 6, c = i & 63;
    int g = row0 + r; if (g > N_GRIDN - 1) g = N_GRIDN - 1;
    *(uint4*)(&bufA[r * A_STRIDE + c * 8]) = *(const uint4*)(&grid_bf[(size_t)g * DD + (size_t)c * 8]);
  }
  for (int i = tid; i < 16 * DD; i += 256) {
    int r = i >> 9, c = i & (DD - 1);
    int g = row0 + r; if (g > N_GRIDN - 1) g = N_GRIDN - 1;
    bufB[r * A_STRIDE + c] = f2bf(agg[(size_t)g * DD + c]);
  }
  __syncthreads();

  // node hidden: silu([grid|agg] @ node_w0 + b0)
  v8f C1[4];
#pragma unroll
  for (int j = 0; j < 4; ++j) C1[j] = splat8(node_b0[(4 * w + j) * 16 + nb]);
  for (int kt = 0; kt < 16; ++kt) {
    const int aoff = nb * A_STRIDE + kt * 32 + (lane >> 4) * 8;
    v16bf a1 = load_a_frag(&bufA[aoff]);
    v16bf a2 = load_a_frag(&bufB[aoff]);
#pragma unroll
    for (int j = 0; j < 4; ++j) {
      long fb = (long)(4 * w + j) * 32;   // node_w0 packed with Kt = 32
      C1[j] = wmma_bf16(a1, load_b_frag(P_node_w0, fb + kt,      lane), C1[j]);
      C1[j] = wmma_bf16(a2, load_b_frag(P_node_w0, fb + 16 + kt, lane), C1[j]);
    }
  }
  __syncthreads();
#pragma unroll
  for (int j = 0; j < 4; ++j) {
    int n = (4 * w + j) * 16 + nb;
#pragma unroll
    for (int r = 0; r < 8; ++r) {
      int m = r + ((lane >> 4) << 3);
      bufA[m * A_STRIDE + n] = f2bf(silu_f(C1[j][r]));
    }
  }
  __syncthreads();

  // node out + residual -> grid_new (bf16 into bufB)
  v8f C2[4];
#pragma unroll
  for (int j = 0; j < 4; ++j) C2[j] = splat8(node_b1[(4 * w + j) * 16 + nb]);
  for (int kt = 0; kt < 16; ++kt) {
    v16bf a = load_a_frag(&bufA[nb * A_STRIDE + kt * 32 + (lane >> 4) * 8]);
#pragma unroll
    for (int j = 0; j < 4; ++j)
      C2[j] = wmma_bf16(a, load_b_frag(P_node_w1, (long)(4 * w + j) * 16 + kt, lane), C2[j]);
  }
#pragma unroll
  for (int j = 0; j < 4; ++j) {
    int n = (4 * w + j) * 16 + nb;
#pragma unroll
    for (int r = 0; r < 8; ++r) {
      int m = r + ((lane >> 4) << 3);
      int g = row0 + m; if (g > N_GRIDN - 1) g = N_GRIDN - 1;
      bufB[m * A_STRIDE + n] = f2bf(C2[j][r] + grid_f32[(size_t)g * DD + n]);
    }
  }
  __syncthreads();

  // output hidden: silu(grid_new @ out_w0 + b0)
  v8f C3[4];
#pragma unroll
  for (int j = 0; j < 4; ++j) C3[j] = splat8(out_b0[(4 * w + j) * 16 + nb]);
  for (int kt = 0; kt < 16; ++kt) {
    v16bf a = load_a_frag(&bufB[nb * A_STRIDE + kt * 32 + (lane >> 4) * 8]);
#pragma unroll
    for (int j = 0; j < 4; ++j)
      C3[j] = wmma_bf16(a, load_b_frag(P_out_w0, (long)(4 * w + j) * 16 + kt, lane), C3[j]);
  }
#pragma unroll
  for (int j = 0; j < 4; ++j) {
    int n = (4 * w + j) * 16 + nb;
#pragma unroll
    for (int r = 0; r < 8; ++r) {
      int m = r + ((lane >> 4) << 3);
      bufA[m * A_STRIDE + n] = f2bf(silu_f(C3[j][r]));
    }
  }
  __syncthreads();

  // final: out = t2 @ out_w1 + b1  (N padded 471 -> 480, Nt = 30)
  v8f C4[4];
#pragma unroll
  for (int j = 0; j < 4; ++j) {
    int n = (4 * w + j) * 16 + nb;
    float bn = ((4 * w + j) < 30 && n < NOUT) ? out_b1[n] : 0.0f;
    C4[j] = splat8(bn);
  }
  for (int kt = 0; kt < 16; ++kt) {
    v16bf a = load_a_frag(&bufA[nb * A_STRIDE + kt * 32 + (lane >> 4) * 8]);
#pragma unroll
    for (int j = 0; j < 4; ++j)
      if ((4 * w + j) < 30)
        C4[j] = wmma_bf16(a, load_b_frag(P_out_w1, (long)(4 * w + j) * 16 + kt, lane), C4[j]);
  }
  int nvalid = N_GRIDN - row0; if (nvalid > 16) nvalid = 16;
#pragma unroll
  for (int j = 0; j < 4; ++j) {
    int nt = 4 * w + j;
    int n = nt * 16 + nb;
#pragma unroll
    for (int r = 0; r < 8; ++r) {
      int m = r + ((lane >> 4) << 3);
      if (nt < 30 && n < NOUT && m < nvalid)
        out[(size_t)(row0 + m) * NOUT + n] = C4[j][r];
    }
  }
}

extern "C" void kernel_launch(void* const* d_in, const int* in_sizes, int n_in,
                              void* d_out, int out_size, void* d_ws, size_t ws_size,
                              hipStream_t stream) {
  (void)in_sizes; (void)n_in; (void)out_size; (void)ws_size;
  const float* mesh    = (const float*)d_in[0];
  const float* gridf   = (const float*)d_in[1];
  const float* eattr   = (const float*)d_in[2];
  const int*   esrc    = (const int*)d_in[3];
  const int*   edst    = (const int*)d_in[4];
  const float* emb_w0  = (const float*)d_in[5];
  const float* emb_b0  = (const float*)d_in[6];
  const float* emb_w1  = (const float*)d_in[7];
  const float* emb_b1  = (const float*)d_in[8];
  const float* edge_w0 = (const float*)d_in[9];
  const float* edge_b0 = (const float*)d_in[10];
  const float* edge_w1 = (const float*)d_in[11];
  const float* edge_b1 = (const float*)d_in[12];
  const float* node_w0 = (const float*)d_in[13];
  const float* node_b0 = (const float*)d_in[14];
  const float* node_w1 = (const float*)d_in[15];
  const float* node_b1 = (const float*)d_in[16];
  const float* out_w0  = (const float*)d_in[17];
  const float* out_b0  = (const float*)d_in[18];
  const float* out_w1  = (const float*)d_in[19];
  const float* out_b1  = (const float*)d_in[20];

  char* ws = (char*)d_ws;
  size_t off = 0;
  auto carve = [&](size_t bytes) {
    void* p = ws + off;
    off = (off + bytes + 255) & ~(size_t)255;
    return p;
  };
  unsigned short* mesh_bf   = (unsigned short*)carve((size_t)N_MESH  * DD * 2);
  unsigned short* grid_bf   = (unsigned short*)carve((size_t)N_GRIDN * DD * 2);
  float*          agg       = (float*)         carve((size_t)N_GRIDN * DD * 4);
  unsigned short* P_emb_w1  = (unsigned short*)carve((size_t)512  * 512 * 2);
  unsigned short* P_edge_w0 = (unsigned short*)carve((size_t)1536 * 512 * 2);
  unsigned short* P_edge_w1 = (unsigned short*)carve((size_t)512  * 512 * 2);
  unsigned short* P_node_w0 = (unsigned short*)carve((size_t)1024 * 512 * 2);
  unsigned short* P_node_w1 = (unsigned short*)carve((size_t)512  * 512 * 2);
  unsigned short* P_out_w0  = (unsigned short*)carve((size_t)512  * 512 * 2);
  unsigned short* P_out_w1  = (unsigned short*)carve((size_t)512  * 480 * 2);

  hipMemsetAsync(agg, 0, (size_t)N_GRIDN * DD * 4, stream);
  k_cast<<<2048, 256, 0, stream>>>(mesh,  mesh_bf, (size_t)N_MESH  * DD);
  k_cast<<<4096, 256, 0, stream>>>(gridf, grid_bf, (size_t)N_GRIDN * DD);

  auto packLaunch = [&](const float* W, unsigned short* P, int K, int Nsrc, int Nt) {
    int threads = (K >> 5) * Nt * 32;
    k_pack<<<(threads + 255) / 256, 256, 0, stream>>>(W, P, K, Nsrc, Nt);
  };
  packLaunch(emb_w1,  P_emb_w1,  512,  512, 32);
  packLaunch(edge_w0, P_edge_w0, 1536, 512, 32);
  packLaunch(edge_w1, P_edge_w1, 512,  512, 32);
  packLaunch(node_w0, P_node_w0, 1024, 512, 32);
  packLaunch(node_w1, P_node_w1, 512,  512, 32);
  packLaunch(out_w0,  P_out_w0,  512,  512, 32);
  packLaunch(out_w1,  P_out_w1,  512,  471, 30);

  k_edge<<<(N_EDGES + 15) / 16, 256, 0, stream>>>(
      mesh_bf, grid_bf, eattr, esrc, edst,
      emb_w0, emb_b0, P_emb_w1, emb_b1,
      P_edge_w0, edge_b0, P_edge_w1, edge_b1, agg);

  k_node<<<(N_GRIDN + 15) / 16, 256, 0, stream>>>(
      grid_bf, gridf, agg,
      P_node_w0, node_b0, P_node_w1, node_b1,
      P_out_w0, out_b0, P_out_w1, out_b1, (float*)d_out);
}
